// EdgeAttrConstruct_32504312496840
// MI455X (gfx1250) — matrix operand order repaired and checked by reading, not compiled, not run on hardware
//
#include <hip/hip_runtime.h>

typedef _Float16 v16h __attribute__((ext_vector_type(16)));
typedef _Float16 v8h  __attribute__((ext_vector_type(8)));
typedef float    v8f  __attribute__((ext_vector_type(8)));
typedef __fp16   pk2_t __attribute__((ext_vector_type(2)));   // cvt_pkrtz result type

#define N_EDGES   800000
#define NBLOCKS   (N_EDGES / 16)
#define SLOPE     0.05f

// LDS weight strides in halves (padded to break bank conflicts, keep 16B align)
#define S0 72    // 64 + 8
#define S1 136   // 128 + 8
#define S2 136
#define W0_OFF 0
#define W1_OFF (128 * S0)            // 9216
#define W2_OFF (W1_OFF + 128 * S1)   // 26624
#define WTOT   (W2_OFF + 64 * S2)    // 35328 halves = 70656 B

// leaky relu with 0 < slope < 1:  max(x, slope*x)  (2 VALU ops, dual-issuable)
__device__ __forceinline__ float lrelu(float v) { return fmaxf(v, SLOPE * v); }

union H2I { pk2_t h; int i; };
union I8V { int i[8]; v16h v; };

__device__ __forceinline__ int pack_lrelu(float a, float b) {
    H2I u;
    u.h = __builtin_amdgcn_cvt_pkrtz(lrelu(a), lrelu(b));
    return u.i;
}

// Build a B fragment (K-window of 32) for the next layer from two 16x16 D tiles
// (packed f16x2 dwords). Lanes 0-15 need K 0..15 of their edge column (tile 2f),
// lanes 16-31 need K 16..31 (tile 2f+1); the "other half" of each tile lives in
// lane^16, fetched with ds_swizzle_b32 SWAPX16 (xor=0x10, and=0x1f -> 0x401f).
__device__ __forceinline__ v16h make_bfrag(const int* pkA, const int* pkB, bool low) {
    I8V u;
#pragma unroll
    for (int j = 0; j < 4; ++j) {
        int sA = __builtin_amdgcn_ds_swizzle(pkA[j], 0x401f);
        int sB = __builtin_amdgcn_ds_swizzle(pkB[j], 0x401f);
        u.i[j]     = low ? pkA[j] : sB;
        u.i[j + 4] = low ? sA     : pkB[j];
    }
    return u.v;
}

// A fragment: 16 contiguous halves K=base..base+15 (matches B fragment K order:
// VGPR0-3 hold K=base..base+7, VGPR4-7 hold K=base+8..base+15).
__device__ __forceinline__ v16h load_afrag(const _Float16* p /*16B aligned*/) {
    v8h lo = *(const v8h*)(p);
    v8h hi = *(const v8h*)(p + 8);
    return __builtin_shufflevector(lo, hi, 0, 1, 2, 3, 4, 5, 6, 7,
                                           8, 9, 10, 11, 12, 13, 14, 15);
}

// ---- kernel 1: transpose + f32->f16 convert weights into d_ws (linear layout)
__global__ __launch_bounds__(256) void prep_weights(const float* __restrict__ W0,
                                                    const float* __restrict__ W1,
                                                    const float* __restrict__ W2,
                                                    _Float16* __restrict__ wt) {
    int i = blockIdx.x * 256 + threadIdx.x;
    if (i < 8192) {                       // W0T[128][64]: n = out-feature, k = in
        int n = i >> 6, k = i & 63;
        wt[i] = (_Float16)W0[k * 128 + n];
    } else if (i < 24576) {               // W1T[128][128]
        int t = i - 8192;
        int n = t >> 7, k = t & 127;
        wt[i] = (_Float16)W1[k * 128 + n];
    } else if (i < 32768) {               // W2T[64][128]
        int t = i - 24576;
        int n = t >> 7, k = t & 127;
        wt[i] = (_Float16)W2[k * 64 + n];
    }
}

// ---- kernel 2: main edge MLP
__global__ __launch_bounds__(256) void edge_mlp_kernel(const float* __restrict__ x,
                                                       const int* __restrict__ ei,
                                                       const _Float16* __restrict__ wsrc,
                                                       const float* __restrict__ b0g,
                                                       const float* __restrict__ b1g,
                                                       const float* __restrict__ b2g,
                                                       float* __restrict__ out) {
    __shared__ __align__(16) _Float16 wt[WTOT];
    __shared__ __align__(16) float bias[320];

    // stage f16 transposed weights into padded LDS layout
    for (int i = threadIdx.x; i < 32768; i += 256) {
        _Float16 v = wsrc[i];
        int dst;
        if (i < 8192)        { int n = i >> 6, k = i & 63;                 dst = W0_OFF + n * S0 + k; }
        else if (i < 24576)  { int t = i - 8192;  int n = t >> 7, k = t & 127; dst = W1_OFF + n * S1 + k; }
        else                 { int t = i - 24576; int n = t >> 7, k = t & 127; dst = W2_OFF + n * S2 + k; }
        wt[dst] = v;
    }
    for (int i = threadIdx.x; i < 320; i += 256)
        bias[i] = (i < 128) ? b0g[i] : (i < 256) ? b1g[i - 128] : b2g[i - 256];
    __syncthreads();

    const int  lane  = threadIdx.x & 31;
    const int  r     = lane & 15;           // edge row within block / weight row
    const bool hiL   = (lane & 16) != 0;
    const int  cbase = hiL ? 8 : 0;         // C/D row base
    const int  kb    = hiL ? 16 : 0;        // B K-base within 32-window
    const int* ei0 = ei;
    const int* ei1 = ei + N_EDGES;

    const int gwave  = blockIdx.x * (blockDim.x >> 5) + (threadIdx.x >> 5);
    const int nwaves = gridDim.x * (blockDim.x >> 5);

#pragma clang loop unroll(disable)
    for (int blk = gwave; blk < NBLOCKS; blk += nwaves) {
        // Opaque zero: makes LDS addresses loop-variant so LICM cannot hoist the
        // 64 loop-invariant weight fragments (512 VGPRs) out of the loop and
        // spill them to scratch. Costs one v_mov_b32 per iteration.
        int izero;
        asm volatile("v_mov_b32 %0, 0" : "=v"(izero));
        const _Float16* wti = wt + izero;
        const float*    bi  = bias + izero;

        const int e = blk * 16 + r;
        const int si = ei0[e], di = ei1[e];
        const float* xs = x + (size_t)si * 64;
        const float* xd = x + (size_t)di * 64;

        // ---- gather e = x[src] + x[dst], convert to f16 B fragments (K=64 -> 2 frags)
        v16h B0f[2];
#pragma unroll
        for (int f = 0; f < 2; ++f) {
            const float4* ps = (const float4*)(xs + f * 32 + kb);
            const float4* pd = (const float4*)(xd + f * 32 + kb);
#pragma unroll
            for (int j = 0; j < 4; ++j) {
                float4 a = ps[j], b = pd[j];
                B0f[f][4 * j + 0] = (_Float16)(a.x + b.x);
                B0f[f][4 * j + 1] = (_Float16)(a.y + b.y);
                B0f[f][4 * j + 2] = (_Float16)(a.z + b.z);
                B0f[f][4 * j + 3] = (_Float16)(a.w + b.w);
            }
        }

        // ---- layer 0: h0 = leaky(W0^T x e^T + b0), 8 output tiles of 16 features
        int pk0[8][4];
#pragma unroll
        for (int mt = 0; mt < 8; ++mt) {
            const _Float16* wrow = wti + W0_OFF + (mt * 16 + r) * S0;
            const float* bp = bi + mt * 16 + cbase;
            float4 c0 = ((const float4*)bp)[0];
            float4 c1 = ((const float4*)bp)[1];
            v8f c = {c0.x, c0.y, c0.z, c0.w, c1.x, c1.y, c1.z, c1.w};
#pragma unroll
            for (int f = 0; f < 2; ++f) {
                v16h a = load_afrag(wrow + f * 32 + kb);
                c = __builtin_amdgcn_wmma_f32_16x16x32_f16(false, a, false, B0f[f],
                                                           (short)0, c, false, false);
            }
#pragma unroll
            for (int j = 0; j < 4; ++j) pk0[mt][j] = pack_lrelu(c[2 * j], c[2 * j + 1]);
        }

        v16h B1f[4];
#pragma unroll
        for (int f = 0; f < 4; ++f) B1f[f] = make_bfrag(pk0[2 * f], pk0[2 * f + 1], !hiL);

        // ---- layer 1: 128x128, K=128 -> 4 fragments per tile
        int pk1[8][4];
#pragma unroll
        for (int mt = 0; mt < 8; ++mt) {
            const _Float16* wrow = wti + W1_OFF + (mt * 16 + r) * S1;
            const float* bp = bi + 128 + mt * 16 + cbase;
            float4 c0 = ((const float4*)bp)[0];
            float4 c1 = ((const float4*)bp)[1];
            v8f c = {c0.x, c0.y, c0.z, c0.w, c1.x, c1.y, c1.z, c1.w};
#pragma unroll
            for (int f = 0; f < 4; ++f) {
                v16h a = load_afrag(wrow + f * 32 + kb);
                c = __builtin_amdgcn_wmma_f32_16x16x32_f16(false, a, false, B1f[f],
                                                           (short)0, c, false, false);
            }
#pragma unroll
            for (int j = 0; j < 4; ++j) pk1[mt][j] = pack_lrelu(c[2 * j], c[2 * j + 1]);
        }

        v16h B2f[4];
#pragma unroll
        for (int f = 0; f < 4; ++f) B2f[f] = make_bfrag(pk1[2 * f], pk1[2 * f + 1], !hiL);

        // ---- layer 2: 128 -> 64, leaky, store f32
#pragma unroll
        for (int mt = 0; mt < 4; ++mt) {
            const _Float16* wrow = wti + W2_OFF + (mt * 16 + r) * S2;
            const float* bp = bi + 256 + mt * 16 + cbase;
            float4 c0 = ((const float4*)bp)[0];
            float4 c1 = ((const float4*)bp)[1];
            v8f c = {c0.x, c0.y, c0.z, c0.w, c1.x, c1.y, c1.z, c1.w};
#pragma unroll
            for (int f = 0; f < 4; ++f) {
                v16h a = load_afrag(wrow + f * 32 + kb);
                c = __builtin_amdgcn_wmma_f32_16x16x32_f16(false, a, false, B2f[f],
                                                           (short)0, c, false, false);
            }
            float* op = out + (size_t)e * 64 + mt * 16 + cbase;
            float4 o0 = make_float4(lrelu(c[0]), lrelu(c[1]), lrelu(c[2]), lrelu(c[3]));
            float4 o1 = make_float4(lrelu(c[4]), lrelu(c[5]), lrelu(c[6]), lrelu(c[7]));
            ((float4*)op)[0] = o0;
            ((float4*)op)[1] = o1;
        }
    }
}

extern "C" void kernel_launch(void* const* d_in, const int* in_sizes, int n_in,
                              void* d_out, int out_size, void* d_ws, size_t ws_size,
                              hipStream_t stream) {
    const float* x  = (const float*)d_in[0];
    const int*   ei = (const int*)d_in[1];
    const float* W0 = (const float*)d_in[2];
    const float* b0 = (const float*)d_in[3];
    const float* W1 = (const float*)d_in[4];
    const float* b1 = (const float*)d_in[5];
    const float* W2 = (const float*)d_in[6];
    const float* b2 = (const float*)d_in[7];
    _Float16* wt16 = (_Float16*)d_ws;

    prep_weights<<<128, 256, 0, stream>>>(W0, W1, W2, wt16);
    edge_mlp_kernel<<<512, 256, 0, stream>>>(x, ei, wt16, b0, b1, b2, (float*)d_out);
}
